// TopologyMessageLayer_90752658964786
// MI455X (gfx1250) — matrix operand order repaired and checked by reading, not compiled, not run on hardware
//
#include <hip/hip_runtime.h>
#include <hip/hip_bf16.h>

// ---------------- problem constants ----------------
#define DIMD   256
#define NFACE  4096
#define NEDGE  8192
#define BATCH  8

typedef __attribute__((ext_vector_type(16))) __bf16 v16bf;
typedef __attribute__((ext_vector_type(8)))  float  v8f;

// ---------------- workspace layout (bytes) ----------------
#define OFF_W1   ((size_t)0)                       // 768*512*2  = 786432
#define OFF_W2   ((size_t)786432)                  // 512*256*2  = 262144
#define OFF_W3   ((size_t)1048576)                 // 512*256*2  = 262144
#define OFF_W4   ((size_t)1310720)                 // 256*256*2  = 131072
#define OFF_SUM  ((size_t)1441792)                 // 8*4096*256*4 = 33554432
#define OFF_CNT  ((size_t)34996224)                // 8*4096*4   = 131072
#define ACC_BYTES ((size_t)(33554432 + 131072))

__device__ __forceinline__ float gelu_exact(float x) {
    return 0.5f * x * (1.0f + erff(x * 0.70710678118654752440f));
}

// A-matrix fragment packing (16x32 bf16 tile, ISA 7.12.2):
//   elems 0..7  : lanes0-15 K=0..7,   lanes16-31 K=8..15
//   elems 8..15 : lanes0-15 K=16..23, lanes16-31 K=24..31
__device__ __forceinline__ int afrag_off(int m, int k) {
    int t    = k >> 5;
    int r    = k & 31;
    int g    = r >> 3;                 // 0..3
    int lane = m + ((g & 1) << 4);
    int elem = ((g >> 1) << 3) + (r & 7);
    return ((t << 5) + lane) * 16 + elem;
}

// ---------------- weight pre-pack: f32 row-major (K,N) -> bf16 B-fragments ----------------
// B-matrix 32x16 bf16 layout: lane holds n = lane&15 ; elems 0..7 : K = (lane<16?0:8)+elem, elems 8..15 : K = 16+(lane<16?0:8)+(elem-8)
__global__ void pack_weight_kernel(const float* __restrict__ W, __bf16* __restrict__ dst,
                                   int K, int N) {
    int e     = blockIdx.x * blockDim.x + threadIdx.x;
    int total = K * N;
    if (e >= total) return;
    int NT = N >> 4;
    int f  = e >> 9;           // frag id (512 elems per frag)
    int l  = (e >> 4) & 31;    // lane
    int i  = e & 15;           // elem
    int kt = f / NT, nt = f - kt * NT;
    int n  = (nt << 4) + (l & 15);
    int k  = (kt << 5) + ((i >> 3) << 4) + ((l >> 4) << 3) + (i & 7);
    dst[e] = (__bf16)W[(size_t)k * N + n];
}

// ---------------- edge kernel: E_new = LN(E + MLP([E,F1,F2])*vf); scatter to face accum ----
__global__ void __launch_bounds__(256)
edge_kernel(const float* __restrict__ E, const float* __restrict__ F,
            const int* __restrict__ etf, const unsigned char* __restrict__ emask,
            const __bf16* __restrict__ bW1, const float* __restrict__ B1,
            const __bf16* __restrict__ bW2, const float* __restrict__ B2,
            const float* __restrict__ GE, const float* __restrict__ BE,
            float* __restrict__ outE, float* __restrict__ faceSum,
            float* __restrict__ faceCnt) {
    __shared__ __align__(32) __bf16 sA[24 * 512];   // A frags, 16x768 bf16
    __shared__ __align__(32) __bf16 sH[16 * 512];   // H frags, 16x512 bf16
    __shared__ float sE[16 * 256];                  // E tile f32 (residual)
    __shared__ float sX[16 * 256];                  // pre-LN result
    __shared__ int   sF1[16], sF2[16];
    __shared__ float sVf[16];

    const int tid = threadIdx.x;
    const int bid = blockIdx.x;
    const int b   = bid >> 9;              // / (NEDGE/16)
    const int e0  = (bid & 511) << 4;

    if (tid < 16) {
        int m    = tid;
        size_t r = (size_t)(b * NEDGE + e0 + m);
        int raw0 = etf[r * 2 + 0];
        int raw1 = etf[r * 2 + 1];
        bool ok  = (emask[r] != 0) && (raw0 >= 0) && (raw1 >= 0);
        sF1[m]   = min(max(raw0, 0), NFACE - 1);
        sF2[m]   = min(max(raw1, 0), NFACE - 1);
        sVf[m]   = ok ? 1.0f : 0.0f;
    }
    __syncthreads();

    // ---- build A = [E | F[f1] | F[f2]] (16 x 768) as bf16 A-fragments ----
    for (int g = tid; g < 16 * 96; g += 256) {
        int m  = g / 96;
        int k0 = (g - m * 96) << 3;
        const float* src;
        if (k0 < 256)      src = E + ((size_t)(b * NEDGE + e0 + m) * DIMD + k0);
        else if (k0 < 512) src = F + ((size_t)(b * NFACE + sF1[m]) * DIMD + (k0 - 256));
        else               src = F + ((size_t)(b * NFACE + sF2[m]) * DIMD + (k0 - 512));
        float4 x0 = *(const float4*)(src);
        float4 x1 = *(const float4*)(src + 4);
        if (k0 < 256) {
            *(float4*)&sE[m * 256 + k0]     = x0;
            *(float4*)&sE[m * 256 + k0 + 4] = x1;
        }
        __align__(16) __bf16 h[8];
        h[0] = (__bf16)x0.x; h[1] = (__bf16)x0.y; h[2] = (__bf16)x0.z; h[3] = (__bf16)x0.w;
        h[4] = (__bf16)x1.x; h[5] = (__bf16)x1.y; h[6] = (__bf16)x1.z; h[7] = (__bf16)x1.w;
        *(uint4*)&sA[afrag_off(m, k0)] = *(const uint4*)h;
    }
    __syncthreads();

    const int w    = tid >> 5;
    const int lane = tid & 31;
    const int hi   = lane >> 4;

    // ---- GEMM1: (16x768) @ (768x512) + b1, GELU -> sH frags ----
    // loop-interchanged: one A-frag load feeds 4 independent accumulators
    {
        v8f acc[4];
        for (int it = 0; it < 4; ++it)
            acc[it] = (v8f){0.f, 0.f, 0.f, 0.f, 0.f, 0.f, 0.f, 0.f};
        for (int t = 0; t < 24; ++t) {
            v16bf a = *(const v16bf*)&sA[(t * 32 + lane) * 16];
            #pragma unroll
            for (int it = 0; it < 4; ++it) {
                int nt   = w + (it << 3);
                v16bf bb = *(const v16bf*)&bW1[(size_t)((t * 32 + nt) * 32 + lane) * 16];
                acc[it] = __builtin_amdgcn_wmma_f32_16x16x32_bf16(false, a, false, bb,
                                                                  (short)0, acc[it],
                                                                  false, false);
            }
        }
        for (int it = 0; it < 4; ++it) {
            int nt   = w + (it << 3);
            int col  = (nt << 4) + (lane & 15);
            float bv = B1[col];
            for (int v = 0; v < 8; ++v) {
                int m = v + (hi << 3);
                sH[afrag_off(m, col)] = (__bf16)gelu_exact(acc[it][v] + bv);
            }
        }
    }
    __syncthreads();

    // ---- GEMM2: (16x512) @ (512x256) + b2; x = E + msg*vf ----
    {
        v8f acc[2];
        acc[0] = (v8f){0.f, 0.f, 0.f, 0.f, 0.f, 0.f, 0.f, 0.f};
        acc[1] = acc[0];
        for (int t = 0; t < 16; ++t) {
            v16bf a = *(const v16bf*)&sH[(t * 32 + lane) * 16];
            #pragma unroll
            for (int it = 0; it < 2; ++it) {
                int nt   = (w << 1) + it;
                v16bf bb = *(const v16bf*)&bW2[(size_t)((t * 16 + nt) * 32 + lane) * 16];
                acc[it] = __builtin_amdgcn_wmma_f32_16x16x32_bf16(false, a, false, bb,
                                                                  (short)0, acc[it],
                                                                  false, false);
            }
        }
        for (int it = 0; it < 2; ++it) {
            int nt   = (w << 1) + it;
            int col  = (nt << 4) + (lane & 15);
            float bv = B2[col];
            for (int v = 0; v < 8; ++v) {
                int m = v + (hi << 3);
                sX[m * 256 + col] = sE[m * 256 + col] + (acc[it][v] + bv) * sVf[m];
            }
        }
    }
    __syncthreads();

    // ---- LayerNorm per row (wave32 shuffle reduce) + store + scatter-add ----
    for (int mi = 0; mi < 2; ++mi) {
        int m = (w << 1) + mi;
        float xs[8], s = 0.f, s2 = 0.f;
        for (int j = 0; j < 8; ++j) {
            float x = sX[m * 256 + lane + (j << 5)];
            xs[j] = x; s += x; s2 += x * x;
        }
        for (int off = 16; off > 0; off >>= 1) {
            s  += __shfl_xor(s,  off, 32);
            s2 += __shfl_xor(s2, off, 32);
        }
        float mean = s * (1.0f / 256.0f);
        float var  = s2 * (1.0f / 256.0f) - mean * mean;
        float rs   = rsqrtf(var + 1e-5f);
        float vf   = sVf[m];
        int   fa   = sF1[m], fb = sF2[m];
        size_t ro  = (size_t)(b * NEDGE + e0 + m) * DIMD;
        size_t ra  = (size_t)(b * NFACE + fa) * DIMD;
        size_t rb  = (size_t)(b * NFACE + fb) * DIMD;
        for (int j = 0; j < 8; ++j) {
            int col  = lane + (j << 5);
            float y  = (xs[j] - mean) * rs * GE[col] + BE[col];
            outE[ro + col] = y;
            float c = y * vf;
            atomicAdd(&faceSum[ra + col], c);
            atomicAdd(&faceSum[rb + col], c);
        }
        if (lane == 0) {
            atomicAdd(&faceCnt[b * NFACE + fa], vf);
            atomicAdd(&faceCnt[b * NFACE + fb], vf);
        }
    }
}

// ---------------- face kernel: F_new = LN(F + MLP([F, face_msg]) * mask) ----------------
__global__ void __launch_bounds__(256)
face_kernel(const float* __restrict__ F, const unsigned char* __restrict__ fmask,
            const __bf16* __restrict__ bW3, const float* __restrict__ B3,
            const __bf16* __restrict__ bW4, const float* __restrict__ B4,
            const float* __restrict__ GF, const float* __restrict__ BF,
            const float* __restrict__ faceSum, const float* __restrict__ faceCnt,
            float* __restrict__ outF) {
    __shared__ __align__(32) __bf16 sA[16 * 512];   // A frags, 16x512 bf16
    __shared__ __align__(32) __bf16 sH[8 * 512];    // H2 frags, 16x256 bf16
    __shared__ float sF[16 * 256];                  // F tile f32 (residual)
    __shared__ float sX[16 * 256];                  // pre-LN result
    __shared__ float sCnt[16];
    __shared__ float sMk[16];

    const int tid = threadIdx.x;
    const int bid = blockIdx.x;
    const int b   = bid >> 8;              // / (NFACE/16)
    const int f0  = (bid & 255) << 4;

    if (tid < 16) {
        int m    = tid;
        size_t r = (size_t)(b * NFACE + f0 + m);
        sCnt[m] = faceCnt[r];
        sMk[m]  = (fmask[r] != 0) ? 1.0f : 0.0f;
    }
    __syncthreads();

    // ---- build A = [F | face_msg] (16 x 512) as bf16 A-fragments ----
    for (int g = tid; g < 16 * 64; g += 256) {
        int m  = g >> 6;
        int k0 = (g & 63) << 3;
        float4 x0, x1;
        if (k0 < 256) {
            const float* src = F + ((size_t)(b * NFACE + f0 + m) * DIMD + k0);
            x0 = *(const float4*)(src);
            x1 = *(const float4*)(src + 4);
            *(float4*)&sF[m * 256 + k0]     = x0;
            *(float4*)&sF[m * 256 + k0 + 4] = x1;
        } else {
            const float* src = faceSum + ((size_t)(b * NFACE + f0 + m) * DIMD + (k0 - 256));
            x0 = *(const float4*)(src);
            x1 = *(const float4*)(src + 4);
            float inv = 1.0f / (sCnt[m] + 1e-8f);
            x0.x *= inv; x0.y *= inv; x0.z *= inv; x0.w *= inv;
            x1.x *= inv; x1.y *= inv; x1.z *= inv; x1.w *= inv;
        }
        __align__(16) __bf16 h[8];
        h[0] = (__bf16)x0.x; h[1] = (__bf16)x0.y; h[2] = (__bf16)x0.z; h[3] = (__bf16)x0.w;
        h[4] = (__bf16)x1.x; h[5] = (__bf16)x1.y; h[6] = (__bf16)x1.z; h[7] = (__bf16)x1.w;
        *(uint4*)&sA[afrag_off(m, k0)] = *(const uint4*)h;
    }
    __syncthreads();

    const int w    = tid >> 5;
    const int lane = tid & 31;
    const int hi   = lane >> 4;

    // ---- GEMM3: (16x512) @ (512x256) + b3, GELU -> sH frags ----
    {
        v8f acc[2];
        acc[0] = (v8f){0.f, 0.f, 0.f, 0.f, 0.f, 0.f, 0.f, 0.f};
        acc[1] = acc[0];
        for (int t = 0; t < 16; ++t) {
            v16bf a = *(const v16bf*)&sA[(t * 32 + lane) * 16];
            #pragma unroll
            for (int it = 0; it < 2; ++it) {
                int nt   = (w << 1) + it;
                v16bf bb = *(const v16bf*)&bW3[(size_t)((t * 16 + nt) * 32 + lane) * 16];
                acc[it] = __builtin_amdgcn_wmma_f32_16x16x32_bf16(false, a, false, bb,
                                                                  (short)0, acc[it],
                                                                  false, false);
            }
        }
        for (int it = 0; it < 2; ++it) {
            int nt   = (w << 1) + it;
            int col  = (nt << 4) + (lane & 15);
            float bv = B3[col];
            for (int v = 0; v < 8; ++v) {
                int m = v + (hi << 3);
                sH[afrag_off(m, col)] = (__bf16)gelu_exact(acc[it][v] + bv);
            }
        }
    }
    __syncthreads();

    // ---- GEMM4: (16x256) @ (256x256) + b4; x = F + upd*mask ----
    {
        v8f acc[2];
        acc[0] = (v8f){0.f, 0.f, 0.f, 0.f, 0.f, 0.f, 0.f, 0.f};
        acc[1] = acc[0];
        for (int t = 0; t < 8; ++t) {
            v16bf a = *(const v16bf*)&sH[(t * 32 + lane) * 16];
            #pragma unroll
            for (int it = 0; it < 2; ++it) {
                int nt   = (w << 1) + it;
                v16bf bb = *(const v16bf*)&bW4[(size_t)((t * 16 + nt) * 32 + lane) * 16];
                acc[it] = __builtin_amdgcn_wmma_f32_16x16x32_bf16(false, a, false, bb,
                                                                  (short)0, acc[it],
                                                                  false, false);
            }
        }
        for (int it = 0; it < 2; ++it) {
            int nt   = (w << 1) + it;
            int col  = (nt << 4) + (lane & 15);
            float bv = B4[col];
            for (int v = 0; v < 8; ++v) {
                int m = v + (hi << 3);
                sX[m * 256 + col] = sF[m * 256 + col] + (acc[it][v] + bv) * sMk[m];
            }
        }
    }
    __syncthreads();

    // ---- LayerNorm per row + store F_new ----
    for (int mi = 0; mi < 2; ++mi) {
        int m = (w << 1) + mi;
        float xs[8], s = 0.f, s2 = 0.f;
        for (int j = 0; j < 8; ++j) {
            float x = sX[m * 256 + lane + (j << 5)];
            xs[j] = x; s += x; s2 += x * x;
        }
        for (int off = 16; off > 0; off >>= 1) {
            s  += __shfl_xor(s,  off, 32);
            s2 += __shfl_xor(s2, off, 32);
        }
        float mean = s * (1.0f / 256.0f);
        float var  = s2 * (1.0f / 256.0f) - mean * mean;
        float rs   = rsqrtf(var + 1e-5f);
        size_t ro  = (size_t)(b * NFACE + f0 + m) * DIMD;
        for (int j = 0; j < 8; ++j) {
            int col = lane + (j << 5);
            outF[ro + col] = (xs[j] - mean) * rs * GF[col] + BF[col];
        }
    }
}

// ---------------- host side ----------------
extern "C" void kernel_launch(void* const* d_in, const int* in_sizes, int n_in,
                              void* d_out, int out_size, void* d_ws, size_t ws_size,
                              hipStream_t stream) {
    (void)in_sizes; (void)n_in; (void)out_size; (void)ws_size;

    const float*         F     = (const float*)d_in[0];
    const float*         E     = (const float*)d_in[1];
    const int*           etf   = (const int*)d_in[2];
    const unsigned char* fmask = (const unsigned char*)d_in[3];
    const unsigned char* emask = (const unsigned char*)d_in[4];
    const float* W1 = (const float*)d_in[5];  const float* b1 = (const float*)d_in[6];
    const float* W2 = (const float*)d_in[7];  const float* b2 = (const float*)d_in[8];
    const float* W3 = (const float*)d_in[9];  const float* b3 = (const float*)d_in[10];
    const float* W4 = (const float*)d_in[11]; const float* b4 = (const float*)d_in[12];
    const float* gf = (const float*)d_in[13]; const float* bf = (const float*)d_in[14];
    const float* ge = (const float*)d_in[15]; const float* be = (const float*)d_in[16];

    char* ws = (char*)d_ws;
    __bf16* bW1  = (__bf16*)(ws + OFF_W1);
    __bf16* bW2  = (__bf16*)(ws + OFF_W2);
    __bf16* bW3  = (__bf16*)(ws + OFF_W3);
    __bf16* bW4  = (__bf16*)(ws + OFF_W4);
    float*  fSum = (float*)(ws + OFF_SUM);
    float*  fCnt = (float*)(ws + OFF_CNT);

    float* outF = (float*)d_out;
    float* outE = outF + (size_t)BATCH * NFACE * DIMD;

    hipMemsetAsync(ws + OFF_SUM, 0, ACC_BYTES, stream);

    pack_weight_kernel<<<dim3(1536), dim3(256), 0, stream>>>(W1, bW1, 768, 512);
    pack_weight_kernel<<<dim3(512),  dim3(256), 0, stream>>>(W2, bW2, 512, 256);
    pack_weight_kernel<<<dim3(512),  dim3(256), 0, stream>>>(W3, bW3, 512, 256);
    pack_weight_kernel<<<dim3(256),  dim3(256), 0, stream>>>(W4, bW4, 256, 256);

    edge_kernel<<<dim3(BATCH * (NEDGE / 16)), dim3(256), 0, stream>>>(
        E, F, etf, emask, bW1, b1, bW2, b2, ge, be, outE, fSum, fCnt);

    face_kernel<<<dim3(BATCH * (NFACE / 16)), dim3(256), 0, stream>>>(
        F, fmask, bW3, b3, bW4, b4, gf, bf, fSum, fCnt, outF);
}
